// TemporalAxialAttention_67070209294664
// MI455X (gfx1250) — compile-verified, matrix operand order checked by reading, not run on hardware
//
#include <hip/hip_runtime.h>
#include <hip/hip_bf16.h>

// ---------------------------------------------------------------------------
// TemporalAxialAttention for MI455X (gfx1250, wave32, WMMA f32<=f16xf16)
//
//   1. x (f32) -> xf16
//   2. w_qkv -> wqkvT f16 [3072][1024], w_out -> woutT f16 [1024][1024]
//   3. QKV GEMM (WMMA, 32x64/wave, ping-pong double buffering, no copies)
//      + RoPE epilogue -> qf/kf/vf f16 in (n,head,t,d)
//   4. attention per (n,head): TDM tensor_load_to_lds Q/K/V, QK^T WMMA,
//      causal softmax, PV WMMA -> attn-out f16 in (b,t,hw,e)
//   5. out projection GEMM (WMMA) + bias -> d_out (f32)
// ---------------------------------------------------------------------------

typedef __attribute__((ext_vector_type(16))) _Float16 v16h;
typedef __attribute__((ext_vector_type(8)))  _Float16 v8h;
typedef __attribute__((ext_vector_type(8)))  float    v8f;
typedef __attribute__((ext_vector_type(4)))  unsigned int v4u;
typedef __attribute__((ext_vector_type(4)))  int      v4i;
typedef __attribute__((ext_vector_type(8)))  int      v8i;

#define N_B     2
#define N_T     64
#define N_HW    256
#define N_D     1024
#define N_HEADS 16
#define N_DH    64
#define N_M     32768
#define N_QKV   3072
#define LN10K_OVER_64 0.14391156831212787f  // ln(10000)/64

#if __has_builtin(__builtin_amdgcn_tensor_load_to_lds)
#define USE_TDM 1
#if __has_include(<hip/amd_detail/amd_gfx1250_TDM.h>)
#define TDM_6ARG 1
#endif
#endif

// ---- f16 fragment loader (CDNA5 16-bit A/B layout, 16 rows x 32 K) -------
__device__ __forceinline__ v16h load_frag16(const _Float16* __restrict__ p0,
                                            int row_base, int kb, int ld) {
  const int lane = threadIdx.x & 31;
  const int hs   = lane >> 4;
  const int m    = lane & 15;
  const _Float16* p = p0 + (row_base + m) * ld + kb + hs * 8;
  v8h lo = *(const v8h*)(p);
  v8h hi = *(const v8h*)(p + 16);
  v16h r;
#pragma unroll
  for (int i = 0; i < 8; ++i) { r[i] = lo[i]; r[i + 8] = hi[i]; }
  return r;
}

__device__ __forceinline__ v8f wmma_f32_f16(v16h a, v16h b, v8f c) {
  return __builtin_amdgcn_wmma_f32_16x16x32_f16(false, a, false, b,
                                                (short)0, c, false, false);
}

#ifdef USE_TDM
// 1-D TDM load: 4096 f16 elements (8KB) global -> LDS.
// D# per CDNA5 ISA ch.8: group0 = {count=1 | lds_addr | global_addr | type=2},
// group1 = {data_size=2B, tensor_dim0=4096, tensor_dim1=1, tile_dim0=4096}.
__device__ __forceinline__ void tdm_load_8kb(unsigned lds_off, const void* gptr) {
  unsigned long long ga = (unsigned long long)gptr;
  v4u g0 = { 1u, lds_off, (unsigned)ga,
             (unsigned)((ga >> 32) & 0x01FFFFFFu) | (2u << 30) };
  v8i g1 = { 0x10000, (int)(4096u << 16), (int)(1u << 16), (int)(4096u << 16),
             0, 0, 0, 0 };
  v4i z4 = { 0, 0, 0, 0 };
#ifdef TDM_6ARG
  v8i z8 = { 0, 0, 0, 0, 0, 0, 0, 0 };
  __builtin_amdgcn_tensor_load_to_lds(g0, g1, z4, z4, z8, 0);
#else
  __builtin_amdgcn_tensor_load_to_lds(g0, g1, z4, z4, 0);
#endif
}
#endif

// ---- conversion / transpose kernels --------------------------------------
__global__ void cvt_f32_to_f16_kernel(const float* __restrict__ in,
                                      _Float16* __restrict__ out, int n) {
  int stride = gridDim.x * blockDim.x;
  for (int i = blockIdx.x * blockDim.x + threadIdx.x; i < n; i += stride)
    out[i] = (_Float16)in[i];
}

__global__ void transpose_w_kernel(const float* __restrict__ src,
                                   _Float16* __restrict__ dst, int K, int N) {
  int tid = blockIdx.x * blockDim.x + threadIdx.x;
  if (tid >= K * N) return;
  int k = tid & (K - 1);
  int n = tid / K;
  dst[n * K + k] = (_Float16)src[(size_t)k * N + n];
}

// ---- shared GEMM core: 32x64 per wave, ping-pong K pipeline --------------
// acc[2][4] over rows [rbase, rbase+32) x cols [nbase, nbase+64), K = 1024.
#define GEMM_LOAD(A_, B_, kb_)                                             \
  do {                                                                     \
    _Pragma("unroll")                                                      \
    for (int i_ = 0; i_ < 2; ++i_)                                         \
      (A_)[i_] = load_frag16(amat, rbase + 16 * i_, (kb_), N_D);           \
    _Pragma("unroll")                                                      \
    for (int j_ = 0; j_ < 4; ++j_)                                         \
      (B_)[j_] = load_frag16(bmat, nbase + 16 * j_, (kb_), N_D);           \
  } while (0)

#define GEMM_MMA(A_, B_)                                                   \
  do {                                                                     \
    _Pragma("unroll")                                                      \
    for (int j_ = 0; j_ < 4; ++j_) {                                       \
      acc[0][j_] = wmma_f32_f16((A_)[0], (B_)[j_], acc[0][j_]);            \
      acc[1][j_] = wmma_f32_f16((A_)[1], (B_)[j_], acc[1][j_]);            \
    }                                                                      \
  } while (0)

#define GEMM_BODY()                                                        \
  v8f acc[2][4] = {};                                                      \
  v16h a0[2], b0[4], a1[2], b1[4];                                         \
  GEMM_LOAD(a0, b0, 0);                                                    \
  _Pragma("unroll 1")                                                      \
  for (int kb = 32; kb <= N_D - 64; kb += 64) {                            \
    GEMM_LOAD(a1, b1, kb);                                                 \
    GEMM_MMA(a0, b0);                                                      \
    GEMM_LOAD(a0, b0, kb + 32);                                            \
    GEMM_MMA(a1, b1);                                                      \
  }                                                                        \
  GEMM_LOAD(a1, b1, N_D - 32);                                             \
  GEMM_MMA(a0, b0);                                                        \
  GEMM_MMA(a1, b1)

// ---- QKV GEMM + RoPE + scatter to (n,head,t,d) ---------------------------
__global__ __launch_bounds__(256)
void qkv_rope_kernel(const _Float16* __restrict__ amat,   // xf16 [32768][1024]
                     const _Float16* __restrict__ bmat,   // wqkvT [3072][1024]
                     _Float16* __restrict__ qf,
                     _Float16* __restrict__ kf,
                     _Float16* __restrict__ vf) {
  const int lane = threadIdx.x & 31;
  const int wid  = threadIdx.x >> 5;      // 8 waves: 2 M x 4 N
  const int mw   = wid & 1, nw = wid >> 1;
  const int rbase = blockIdx.x * 64  + mw * 32;
  const int nbase = blockIdx.y * 256 + nw * 64;

  GEMM_BODY();

  const int hs = lane >> 4, m16 = lane & 15;
#pragma unroll
  for (int mi = 0; mi < 2; ++mi) {
#pragma unroll
    for (int j = 0; j < 4; ++j) {
      const int e    = nbase + 16 * j + m16;
      const int part = e >> 10;                 // 0:q 1:k 2:v (wave-uniform)
      const int head = (e >> 6) & 15;
      const int d    = e & 63;
      const float invf = __expf(-LN10K_OVER_64 * (float)(d & ~1));
#pragma unroll
      for (int r = 0; r < 8; ++r) {
        const int row = rbase + 16 * mi + r + 8 * hs;
        const int b_  = row >> 14;
        const int t   = (row >> 8) & 63;
        const int hw  = row & 255;
        float val     = acc[mi][j][r];
        float partner = __shfl_xor(val, 1, 32);   // interleaved RoPE pair
        size_t idx = ((size_t)((b_ * N_HW + hw) * N_HEADS + head) * N_T + t)
                     * N_DH + d;
        if (part == 2) {
          vf[idx] = (_Float16)val;
        } else {
          float ang = (float)t * invf;
          float c = __cosf(ang), s = __sinf(ang);
          float res = (d & 1) ? (val * c + partner * s)
                              : (val * c - partner * s);
          if (part == 0) qf[idx] = (_Float16)res;
          else           kf[idx] = (_Float16)res;
        }
      }
    }
  }
}

// ---- fused causal attention per (n,head) ---------------------------------
__global__ __launch_bounds__(128)
void attn_kernel(const _Float16* __restrict__ qf,
                 const _Float16* __restrict__ kf,
                 const _Float16* __restrict__ vf,
                 _Float16* __restrict__ ao) {   // (b,t,hw,e) f16
  __shared__ _Float16 Qs[N_T * N_DH];
  __shared__ _Float16 Ks[N_T * N_DH];
  __shared__ _Float16 Vt[N_DH * N_T];           // V transposed: Vt[d][s]
  __shared__ _Float16 Ps[4][16 * N_T];          // per-wave prob tiles

  const int tid  = threadIdx.x;
  const int lane = tid & 31;
  const int wave = tid >> 5;
  const int nh   = blockIdx.x;
  const size_t base = (size_t)nh * (N_T * N_DH);

#ifdef USE_TDM
  __shared__ _Float16 Vs[N_T * N_DH];
  if (wave == 0) {                      // one wave issues the 3 DMA tiles
    tdm_load_8kb((unsigned)(size_t)(void*)Qs, qf + base);
    tdm_load_8kb((unsigned)(size_t)(void*)Ks, kf + base);
    tdm_load_8kb((unsigned)(size_t)(void*)Vs, vf + base);
    __builtin_amdgcn_s_wait_tensorcnt(0);
  }
  __syncthreads();
  for (int g = tid; g < (N_T * N_DH) / 8; g += 128) {  // transpose V in LDS
    v8h vv = ((const v8h*)Vs)[g];
    int s = g >> 3, d0 = (g & 7) * 8;
#pragma unroll
    for (int i = 0; i < 8; ++i) Vt[(d0 + i) * N_T + s] = vv[i];
  }
#else
  const v8h* qg = (const v8h*)(qf + base);
  const v8h* kg = (const v8h*)(kf + base);
  const v8h* vg = (const v8h*)(vf + base);
  for (int g = tid; g < (N_T * N_DH) / 8; g += 128) {
    ((v8h*)Qs)[g] = qg[g];
    ((v8h*)Ks)[g] = kg[g];
    v8h vv = vg[g];
    int s = g >> 3, d0 = (g & 7) * 8;
#pragma unroll
    for (int i = 0; i < 8; ++i) Vt[(d0 + i) * N_T + s] = vv[i];
  }
#endif
  __syncthreads();

  const int tb  = wave * 16;
  const int hs  = lane >> 4, m16 = lane & 15;
  v16h aQ0 = load_frag16(Qs, tb, 0,  N_DH);
  v16h aQ1 = load_frag16(Qs, tb, 32, N_DH);

  float p[4][8];
#pragma unroll
  for (int j = 0; j < 4; ++j) {
    v8f accS = {};
    accS = wmma_f32_f16(aQ0, load_frag16(Ks, 16 * j, 0,  N_DH), accS);
    accS = wmma_f32_f16(aQ1, load_frag16(Ks, 16 * j, 32, N_DH), accS);
#pragma unroll
    for (int r = 0; r < 8; ++r) {
      int t = tb + r + 8 * hs;
      int s = 16 * j + m16;
      p[j][r] = (s <= t) ? accS[r] * 0.125f : -1.0e30f;  // scale + causal
    }
  }

#pragma unroll
  for (int r = 0; r < 8; ++r) {
    float rm = fmaxf(fmaxf(p[0][r], p[1][r]), fmaxf(p[2][r], p[3][r]));
#pragma unroll
    for (int off = 1; off < 16; off <<= 1) rm = fmaxf(rm, __shfl_xor(rm, off, 32));
    float rs = 0.f;
#pragma unroll
    for (int j = 0; j < 4; ++j) { p[j][r] = __expf(p[j][r] - rm); rs += p[j][r]; }
#pragma unroll
    for (int off = 1; off < 16; off <<= 1) rs += __shfl_xor(rs, off, 32);
    float ri = 1.0f / rs;
#pragma unroll
    for (int j = 0; j < 4; ++j) p[j][r] *= ri;
  }

  _Float16* Pw = &Ps[wave][0];
#pragma unroll
  for (int j = 0; j < 4; ++j)
#pragma unroll
    for (int r = 0; r < 8; ++r)
      Pw[(r + 8 * hs) * N_T + 16 * j + m16] = (_Float16)p[j][r];
  __syncthreads();

  v16h aP0 = load_frag16(Pw, 0, 0,  N_T);
  v16h aP1 = load_frag16(Pw, 0, 32, N_T);

  const int head = nh & 15, n = nh >> 4;
  const int b_ = n >> 8, hw = n & 255;
#pragma unroll
  for (int j = 0; j < 4; ++j) {
    v8f accO = {};
    accO = wmma_f32_f16(aP0, load_frag16(Vt, 16 * j, 0,  N_T), accO);
    accO = wmma_f32_f16(aP1, load_frag16(Vt, 16 * j, 32, N_T), accO);
#pragma unroll
    for (int r = 0; r < 8; ++r) {
      int t = tb + r + 8 * hs;
      int d = 16 * j + m16;
      size_t dst = ((size_t)(b_ * N_T + t) * N_HW + hw) * N_D + head * N_DH + d;
      ao[dst] = (_Float16)accO[r];
    }
  }
}

// ---- output projection GEMM + bias ---------------------------------------
__global__ __launch_bounds__(256)
void proj_kernel(const _Float16* __restrict__ amat,   // attn-out [32768][1024]
                 const _Float16* __restrict__ bmat,   // woutT [1024][1024]
                 const float* __restrict__ bias,
                 float* __restrict__ out) {
  const int lane = threadIdx.x & 31;
  const int wid  = threadIdx.x >> 5;
  const int mw   = wid & 1, nw = wid >> 1;
  const int rbase = blockIdx.x * 64  + mw * 32;
  const int nbase = blockIdx.y * 256 + nw * 64;

  GEMM_BODY();

  const int hs = lane >> 4, m16 = lane & 15;
#pragma unroll
  for (int mi = 0; mi < 2; ++mi)
#pragma unroll
    for (int j = 0; j < 4; ++j) {
      int col = nbase + 16 * j + m16;
      float bv = bias[col];
#pragma unroll
      for (int r = 0; r < 8; ++r) {
        int row = rbase + 16 * mi + r + 8 * hs;
        out[(size_t)row * N_D + col] = acc[mi][j][r] + bv;
      }
    }
}

// ---------------------------------------------------------------------------
extern "C" void kernel_launch(void* const* d_in, const int* in_sizes, int n_in,
                              void* d_out, int out_size, void* d_ws, size_t ws_size,
                              hipStream_t stream) {
  const float* x     = (const float*)d_in[0];
  const float* w_qkv = (const float*)d_in[1];
  const float* w_out = (const float*)d_in[2];
  const float* b_out = (const float*)d_in[3];
  float* out = (float*)d_out;

  char* w = (char*)d_ws;
  _Float16* xf16  = (_Float16*)(w);                        //  64 MB
  _Float16* wqkvT = (_Float16*)(w + 67108864);             //   6 MB
  _Float16* woutT = (_Float16*)(w + 73400320);             //   2 MB
  _Float16* qf    = (_Float16*)(w + 75497472);             //  64 MB
  _Float16* kf    = (_Float16*)(w + 142606336);            //  64 MB
  _Float16* vf    = (_Float16*)(w + 209715200);            //  64 MB
  _Float16* ao    = xf16;   // attn-out reuses x-f16 region after QKV GEMM

  cvt_f32_to_f16_kernel<<<8192, 256, 0, stream>>>(x, xf16, N_M * N_D);

  transpose_w_kernel<<<(N_D * N_QKV + 255) / 256, 256, 0, stream>>>(
      w_qkv, wqkvT, N_D, N_QKV);
  transpose_w_kernel<<<(N_D * N_D + 255) / 256, 256, 0, stream>>>(
      w_out, woutT, N_D, N_D);

  qkv_rope_kernel<<<dim3(N_M / 64, N_QKV / 256), 256, 0, stream>>>(
      xf16, wqkvT, qf, kf, vf);

  attn_kernel<<<N_B * N_HW * N_HEADS, 128, 0, stream>>>(qf, kf, vf, ao);

  proj_kernel<<<dim3(N_M / 64, N_D / 256), 256, 0, stream>>>(
      ao, woutT, b_out, out);
}
